// HomoGuideAttention_3023656976709
// MI455X (gfx1250) — matrix operand order repaired and checked by reading, not compiled, not run on hardware
//
#include <hip/hip_runtime.h>
#include <hip/hip_bf16.h>

// ---- problem constants (from reference setup_inputs) ----
#define B_  4
#define N_  6
#define D_  128
#define H_  128
#define W_  128
#define P_  (H_ * W_)         // 16384
#define M_  (B_ * P_)         // 65536 pixel-rows
#define DH_ 256               // MLP hidden

typedef __attribute__((ext_vector_type(16))) __bf16 v16bf;
typedef __attribute__((ext_vector_type(8)))  __bf16 v8bf;
typedef __attribute__((ext_vector_type(8)))  float  v8f;

__device__ __forceinline__ __bf16 f2bf(float f) {
  // round-to-nearest-even fp32 -> bf16
  unsigned u = __float_as_uint(f);
  unsigned r = (u + 0x7fffu + ((u >> 16) & 1u)) >> 16;
  unsigned short s = (unsigned short)r;
  __bf16 o;
  __builtin_memcpy(&o, &s, 2);
  return o;
}

// ---------------- kernel 0: pack weights fp32 -> bf16 in WMMA B-fragment order ----
// Packed layout: frag[((kt*NT + nt)*32 + lane)*16 + e] = W[k*N + n]
//   with k = kt*32 + (lane>>4)*8 + (e<8 ? e : 8+e), n = nt*16 + (lane&15)
// so each lane's 16 B-elements are 32 contiguous bytes (two b128 loads).
__global__ void k_pack_weights(const float* __restrict__ w1,
                               const float* __restrict__ w2,
                               __bf16* __restrict__ w1p,
                               __bf16* __restrict__ w2p) {
  int i = blockIdx.x * blockDim.x + threadIdx.x;   // 2 * 32768 elements
  const float* src;
  __bf16* dst;
  int N, NT, j;
  if (i < D_ * DH_) {            // W1: K=128, N=256
    src = w1; dst = w1p; N = DH_; NT = DH_ / 16; j = i;
  } else {                       // W2: K=256, N=128
    src = w2; dst = w2p; N = D_;  NT = D_ / 16;  j = i - D_ * DH_;
  }
  int e    = j & 15;
  int lane = (j >> 4) & 31;
  int tile = j >> 9;
  int nt = tile % NT, kt = tile / NT;
  int k = kt * 32 + (lane >> 4) * 8 + (e < 8 ? e : 8 + e);
  int n = nt * 16 + (lane & 15);
  dst[j] = f2bf(src[(size_t)k * N + n]);
}

// ---------------- kernel 1: 1/max(||q||,eps) per pixel ----------------
__global__ void k_qnorm(const float* __restrict__ query, float* __restrict__ qinv) {
  int gid = blockIdx.x * blockDim.x + threadIdx.x;   // over M_
  int b = gid / P_, p = gid % P_;
  const float* qp = query + (size_t)b * D_ * P_ + p;
  float ss = 0.f;
  #pragma unroll 4
  for (int d = 0; d < D_; ++d) { float q = *qp; qp += P_; ss += q * q; }
  qinv[gid] = 1.0f / fmaxf(sqrtf(ss), 1e-12f);
}

// ---------------- kernel 2: bilinear + cosine-sim dot (pre-softmax) ----------------
__global__ void k_dot(const float* __restrict__ query, const float* __restrict__ ref,
                      const float* __restrict__ huv, const float* __restrict__ qinv,
                      float* __restrict__ dotout) {
  int gid = blockIdx.x * blockDim.x + threadIdx.x;   // over B_*N_*P_
  int b   = gid / (N_ * P_);
  int rem = gid % (N_ * P_);
  int n = rem / P_, p = rem % P_;

  size_t uvb = ((size_t)(b * N_ + n)) * 2 * P_;
  float x = huv[uvb + p] * (float)W_;
  float y = huv[uvb + P_ + p] * (float)H_;
  float x0f = floorf(x), y0f = floorf(y);
  float wx = x - x0f, wy = y - y0f;
  int x0 = (int)x0f, y0 = (int)y0f;
  float valid = (x >= 0.f && x <= (float)(W_ - 1) && y >= 0.f && y <= (float)(H_ - 1)) ? 1.f : 0.f;
  int x0c = min(max(x0, 0), W_ - 1), x1c = min(max(x0 + 1, 0), W_ - 1);
  int y0c = min(max(y0, 0), H_ - 1), y1c = min(max(y0 + 1, 0), H_ - 1);
  int i00 = y0c * W_ + x0c, i10 = y0c * W_ + x1c;
  int i01 = y1c * W_ + x0c, i11 = y1c * W_ + x1c;
  float w00 = (1.f - wx) * (1.f - wy), w10 = wx * (1.f - wy);
  float w01 = (1.f - wx) * wy,          w11 = wx * wy;

  const float* rp = ref + ((size_t)(b * N_ + n)) * D_ * P_;
  const float* qp = query + (size_t)b * D_ * P_ + p;
  float dt = 0.f, ss = 0.f;
  for (int d = 0; d < D_; ++d) {
    float v = w00 * rp[i00] + w10 * rp[i10] + w01 * rp[i01] + w11 * rp[i11];
    float q = *qp;
    qp += P_; rp += P_;
    dt += q * v;
    ss += v * v;
  }
  float kinv = 1.0f / fmaxf(sqrtf(ss), 1e-12f);
  dotout[gid] = dt * qinv[b * P_ + p] * kinv * valid;
}

// ---------------- kernel 3: softmax(n) + weighted gather + residual + LN1 ----------------
__global__ void k_attn_ln1(const float* __restrict__ query, const float* __restrict__ ref,
                           const float* __restrict__ huv, const float* __restrict__ dotin,
                           const float* __restrict__ ln1w, const float* __restrict__ ln1b,
                           float* __restrict__ ztf, __bf16* __restrict__ ztb) {
  int gid = blockIdx.x * blockDim.x + threadIdx.x;   // over M_
  int b = gid / P_, p = gid % P_;

  // softmax over the 6 views
  float dv[N_];
  float mx = -1e30f;
  #pragma unroll
  for (int n = 0; n < N_; ++n) {
    dv[n] = dotin[((size_t)(b * N_ + n)) * P_ + p];
    mx = fmaxf(mx, dv[n]);
  }
  float s = 0.f;
  #pragma unroll
  for (int n = 0; n < N_; ++n) { dv[n] = expf(dv[n] - mx); s += dv[n]; }
  float inv = 1.f / s;

  // corner indices + attention-folded bilinear weights per view
  int   j00[N_], j10[N_], j01[N_], j11[N_];
  float c00[N_], c10[N_], c01[N_], c11[N_];
  const float* rp[N_];
  #pragma unroll
  for (int n = 0; n < N_; ++n) {
    size_t uvb = ((size_t)(b * N_ + n)) * 2 * P_;
    float x = huv[uvb + p] * (float)W_;
    float y = huv[uvb + P_ + p] * (float)H_;
    float x0f = floorf(x), y0f = floorf(y);
    float wx = x - x0f, wy = y - y0f;
    int x0 = (int)x0f, y0 = (int)y0f;
    int x0c = min(max(x0, 0), W_ - 1), x1c = min(max(x0 + 1, 0), W_ - 1);
    int y0c = min(max(y0, 0), H_ - 1), y1c = min(max(y0 + 1, 0), H_ - 1);
    j00[n] = y0c * W_ + x0c; j10[n] = y0c * W_ + x1c;
    j01[n] = y1c * W_ + x0c; j11[n] = y1c * W_ + x1c;
    float att = dv[n] * inv;
    c00[n] = att * (1.f - wx) * (1.f - wy);
    c10[n] = att * wx * (1.f - wy);
    c01[n] = att * (1.f - wx) * wy;
    c11[n] = att * wx * wy;
    rp[n] = ref + ((size_t)(b * N_ + n)) * D_ * P_;
  }

  const float* qp = query + (size_t)b * D_ * P_ + p;
  float* zrow = ztf + (size_t)gid * D_;
  float sum = 0.f, sumsq = 0.f;
  for (int d = 0; d < D_; ++d) {
    float z = *qp; qp += P_;
    #pragma unroll
    for (int n = 0; n < N_; ++n) {
      const float* pl = rp[n];
      z += c00[n] * pl[j00[n]] + c10[n] * pl[j10[n]]
         + c01[n] * pl[j01[n]] + c11[n] * pl[j11[n]];
      rp[n] += P_;
    }
    zrow[d] = z;
    sum += z; sumsq += z * z;
  }
  float mu  = sum * (1.f / D_);
  float var = sumsq * (1.f / D_) - mu * mu;
  float rs  = rsqrtf(var + 1e-5f);
  __bf16* zb = ztb + (size_t)gid * D_;
  for (int d = 0; d < D_; ++d) {
    float v = (zrow[d] - mu) * rs * ln1w[d] + ln1b[d];
    zrow[d] = v;            // fp32 copy: MLP residual
    zb[d]   = f2bf(v);      // bf16 copy: WMMA operand
  }
}

// ---- WMMA fragment helpers ----
// A fragment (16-bit A 16x32 layout per CDNA5 ISA 7.12.2): lane hi-half holds
// K+8..15 / K+24..31; both halves are contiguous 16B chunks -> two b128 loads.
__device__ __forceinline__ v16bf load_a_frag(const __bf16* __restrict__ arow, int k0, int hi) {
  v8bf lo = *(const v8bf*)(arow + k0 + hi * 8);
  v8bf hv = *(const v8bf*)(arow + k0 + 16 + hi * 8);
  return __builtin_shufflevector(lo, hv, 0,1,2,3,4,5,6,7,8,9,10,11,12,13,14,15);
}

// B fragment from the pre-packed weight buffer: 32 contiguous bytes per lane.
__device__ __forceinline__ v16bf load_b_frag_packed(const __bf16* __restrict__ Wp,
                                                    int kt, int nt, int NT, int lane) {
  return *(const v16bf*)(Wp + ((size_t)(kt * NT + nt) * 32 + lane) * 16);
}

// ---------------- kernel 4: GEMM1 (M_x128 @ 128x256) + bias + exact GELU ----------------
__global__ void __launch_bounds__(256)
k_gemm1(const __bf16* __restrict__ X, const __bf16* __restrict__ Wp,
        const float* __restrict__ bias, __bf16* __restrict__ Hout) {
  const int K = D_, N = DH_, NT = DH_ / 16;
  int wave = threadIdx.x >> 5, lane = threadIdx.x & 31;
  int lm = lane & 15, hi = lane >> 4;
  int m0 = blockIdx.x * 128 + wave * 16;   // 8 waves * 16 rows
  int nt0 = blockIdx.y * 4;                // 4 col-tiles per wave

  v8f acc[4] = {};
  const __bf16* arow = X + (size_t)(m0 + lm) * K;
  #pragma unroll
  for (int k0 = 0; k0 < K; k0 += 32) {
    v16bf a = load_a_frag(arow, k0, hi);
    #pragma unroll
    for (int t = 0; t < 4; ++t) {
      v16bf bf = load_b_frag_packed(Wp, k0 >> 5, nt0 + t, NT, lane);
      acc[t] = __builtin_amdgcn_wmma_f32_16x16x32_bf16(
          false, a, false, bf, (short)0, acc[t], false, false);
    }
  }
  #pragma unroll
  for (int t = 0; t < 4; ++t) {
    int n = (nt0 + t) * 16 + lm;
    float bv = bias[n];
    #pragma unroll
    for (int r = 0; r < 8; ++r) {
      int m = m0 + hi * 8 + r;                 // C/D layout: VGPR r, hi-half -> M+8
      float v = acc[t][r] + bv;
      v = 0.5f * v * (1.0f + erff(v * 0.70710678118f));   // exact GELU
      Hout[(size_t)m * N + n] = f2bf(v);
    }
  }
}

// ---------------- kernel 5: GEMM2 (M_x256 @ 256x128) + bias + residual + LN2 + transpose ----------------
__global__ void __launch_bounds__(256)
k_gemm2_ln2(const __bf16* __restrict__ Hin, const __bf16* __restrict__ Wp,
            const float* __restrict__ bias, const float* __restrict__ zt,
            const float* __restrict__ ln2w, const float* __restrict__ ln2b,
            float* __restrict__ out) {
  const int K = DH_, NT = D_ / 16;
  __shared__ float zs[64 * 132];            // 64 rows x full 128 cols, padded stride
  int wave = threadIdx.x >> 5, lane = threadIdx.x & 31;
  int lm = lane & 15, hi = lane >> 4;
  int row0 = blockIdx.x * 64;
  int mt = wave & 3;                        // 4 row-tiles
  int ntb = (wave >> 2) * 4;                // 2 column halves, 4 n-tiles each
  int m0 = row0 + mt * 16;

  v8f acc[4] = {};
  const __bf16* arow = Hin + (size_t)(m0 + lm) * K;
  #pragma unroll
  for (int k0 = 0; k0 < K; k0 += 32) {
    v16bf a = load_a_frag(arow, k0, hi);
    #pragma unroll
    for (int t = 0; t < 4; ++t) {
      v16bf bf = load_b_frag_packed(Wp, k0 >> 5, ntb + t, NT, lane);
      acc[t] = __builtin_amdgcn_wmma_f32_16x16x32_bf16(
          false, a, false, bf, (short)0, acc[t], false, false);
    }
  }
  #pragma unroll
  for (int t = 0; t < 4; ++t) {
    int n = (ntb + t) * 16 + lm;
    float bv = bias[n];
    #pragma unroll
    for (int r = 0; r < 8; ++r) {
      int ml = mt * 16 + hi * 8 + r;
      zs[ml * 132 + n] = acc[t][r] + bv + zt[(size_t)(row0 + ml) * D_ + n];
    }
  }
  __syncthreads();

  int tid = threadIdx.x;
  if (tid < 64) {
    const float* zr = zs + tid * 132;
    float sum = 0.f, ss = 0.f;
    #pragma unroll 4
    for (int d = 0; d < D_; ++d) { float v = zr[d]; sum += v; ss += v * v; }
    float mu  = sum * (1.f / D_);
    float var = ss * (1.f / D_) - mu * mu;
    float rs  = rsqrtf(var + 1e-5f);
    int grow = row0 + tid;
    int b = grow / P_, p = grow % P_;
    float* ob = out + (size_t)b * D_ * P_ + p;
    for (int d = 0; d < D_; ++d) {
      ob[(size_t)d * P_] = (zr[d] - mu) * rs * ln2w[d] + ln2b[d];
    }
  }
}

extern "C" void kernel_launch(void* const* d_in, const int* in_sizes, int n_in,
                              void* d_out, int out_size, void* d_ws, size_t ws_size,
                              hipStream_t stream) {
  (void)in_sizes; (void)n_in; (void)out_size; (void)ws_size;
  const float* query = (const float*)d_in[0];
  const float* ref   = (const float*)d_in[1];
  const float* huv   = (const float*)d_in[2];
  const float* ln1w  = (const float*)d_in[3];
  const float* ln1b  = (const float*)d_in[4];
  const float* ln2w  = (const float*)d_in[5];
  const float* ln2b  = (const float*)d_in[6];
  const float* w1    = (const float*)d_in[7];
  const float* b1    = (const float*)d_in[8];
  const float* w2    = (const float*)d_in[9];
  const float* b2    = (const float*)d_in[10];
  float* out = (float*)d_out;

  // workspace carve-up (~86 MB total)
  char* ws = (char*)d_ws;
  size_t off = 0;
  auto take = [&](size_t bytes) -> void* {
    off = (off + 255) & ~(size_t)255;
    void* p = ws + off;
    off += bytes;
    return p;
  };
  float*  qinv = (float*)take(sizeof(float) * (size_t)M_);
  float*  dt   = (float*)take(sizeof(float) * (size_t)B_ * N_ * P_);
  float*  ztf  = (float*)take(sizeof(float) * (size_t)M_ * D_);
  __bf16* ztb  = (__bf16*)take(2ull * M_ * D_);
  __bf16* hb   = (__bf16*)take(2ull * M_ * DH_);
  __bf16* w1p  = (__bf16*)take(2ull * D_ * DH_);
  __bf16* w2p  = (__bf16*)take(2ull * DH_ * D_);

  k_pack_weights<<<(2 * D_ * DH_) / 256, 256, 0, stream>>>(w1, w2, w1p, w2p);
  k_qnorm<<<M_ / 256, 256, 0, stream>>>(query, qinv);
  k_dot<<<(B_ * N_ * P_) / 256, 256, 0, stream>>>(query, ref, huv, qinv, dt);
  k_attn_ln1<<<M_ / 256, 256, 0, stream>>>(query, ref, huv, dt, ln1w, ln1b, ztf, ztb);
  k_gemm1<<<dim3(M_ / 128, DH_ / 64), 256, 0, stream>>>(ztb, w1p, b1, hb);
  k_gemm2_ln2<<<M_ / 64, 256, 0, stream>>>(hb, w2p, b2, ztf, ln2w, ln2b, out);
}
